// ShuffleFormerLayer_41420664603302
// MI455X (gfx1250) — matrix operand order repaired and checked by reading, not compiled, not run on hardware
//
#include <hip/hip_runtime.h>
#include <hip/hip_bf16.h>
#include <stdint.h>

// Problem constants (match reference)
#define BB      4
#define SS      4096
#define DD      1024
#define SDIM    128
#define HH      16
#define NBUCKET 32
#define BSZ     128
#define HDIM    64
#define MM      (BB*SS)   // 16384 rows total

typedef __bf16 bf16;
typedef __attribute__((ext_vector_type(16))) bf16  bf16x16;
typedef __attribute__((ext_vector_type(8)))  bf16  bf16x8;
typedef __attribute__((ext_vector_type(8)))  float f32x8;

#define F32X8_ZERO f32x8{0.f,0.f,0.f,0.f,0.f,0.f,0.f,0.f}

__device__ __forceinline__ bf16 f2bf(float f) {
  uint32_t u = __builtin_bit_cast(uint32_t, f);
  u += 0x7fffu + ((u >> 16) & 1u);            // round-to-nearest-even
  return __builtin_bit_cast(bf16, (uint16_t)(u >> 16));
}

__device__ __forceinline__ f32x8 wmma_bf16(bf16x16 a, bf16x16 b, f32x8 c) {
  // v_wmma_f32_16x16x32_bf16 : D = A(16x32) * B(32x16) + C
  return __builtin_amdgcn_wmma_f32_16x16x32_bf16(false, a, false, b, (short)0, c, false, false);
}

// ---- CDNA5 async global->LDS copy (ASYNCcnt-tracked, no VGPR round trip) ----
// Per-lane: LDS[vdst] = MEM[vaddr], 16 bytes. Generic LDS pointer low 32 bits
// are the LDS byte offset (flat->LDS truncation, ISA 10.2).
__device__ __forceinline__ void async_ld16(uint32_t lds_off, const void* g) {
  asm volatile("global_load_async_to_lds_b128 %0, %1, off"
               :: "v"(lds_off), "v"((uint64_t)(uintptr_t)g) : "memory");
}
__device__ __forceinline__ void wait_async0() {
  asm volatile("s_wait_asynccnt 0" ::: "memory");
}

// ---- WMMA fragment loaders (CDNA5 ISA 7.12.2 layouts) -----------------------
// A fragment: 16x32 (MxK) tile from row-major storage, row stride ld (elems).
__device__ __forceinline__ bf16x16 frag_a(const bf16* base, int ld) {
  int lane = threadIdx.x & 31;
  int m = lane & 15, h = lane >> 4;
  const bf16* p = base + m * ld + h * 8;
  bf16x8 lo = *(const bf16x8*)(p);
  bf16x8 hi = *(const bf16x8*)(p + 16);
  return __builtin_shufflevector(lo, hi, 0,1,2,3,4,5,6,7,8,9,10,11,12,13,14,15);
}
// B fragment: 32x16 (KxN) from K-major storage BT[n][k], row stride ld.
__device__ __forceinline__ bf16x16 frag_b(const bf16* base, int ld) {
  int lane = threadIdx.x & 31;
  int n = lane & 15, h = lane >> 4;
  const bf16* p = base + n * ld + h * 8;
  bf16x8 lo = *(const bf16x8*)(p);
  bf16x8 hi = *(const bf16x8*)(p + 16);
  return __builtin_shufflevector(lo, hi, 0,1,2,3,4,5,6,7,8,9,10,11,12,13,14,15);
}

// =============================================================================
// Kernel 1: C(bf16)[M,N] = A(f32)[M,K] @ W(f32)[K,N] + bias
// block = 256 thr (8 waves), tile 128x128, wave tile 32x64, K step 32.
// (A/W are f32 so staging converts on the VALU path; no async copy here.)
// =============================================================================
__global__ void gemm_bias_bf16(const float* __restrict__ A, const float* __restrict__ W,
                               const float* __restrict__ bias, bf16* __restrict__ C,
                               int Mdim, int Ndim, int Kdim) {
  extern __shared__ char smem[];
  bf16* As = (bf16*)smem;                 // [128][32]
  bf16* WT = (bf16*)(smem + 128*32*2);    // [128 n][32 k]  (K-major)
  const int t = threadIdx.x;
  const int lane = t & 31, wid = t >> 5;
  const int n15 = lane & 15, hf = lane >> 4;
  const int mbase = blockIdx.x * 128;
  const int nbase = blockIdx.y * 128;
  const int wm = (wid >> 1) * 32;
  const int wn = (wid & 1) * 64;

  f32x8 acc[2][4];
#pragma unroll
  for (int mt = 0; mt < 2; ++mt)
#pragma unroll
    for (int nt = 0; nt < 4; ++nt) acc[mt][nt] = F32X8_ZERO;

  for (int k0 = 0; k0 < Kdim; k0 += 32) {
#pragma unroll
    for (int e = 0; e < 16; ++e) {          // stage A: 128x32
      int idx = t * 16 + e;
      int m = idx >> 5, kk = idx & 31;
      As[idx] = f2bf(A[(size_t)(mbase + m) * Kdim + k0 + kk]);
    }
#pragma unroll
    for (int e = 0; e < 16; ++e) {          // stage W transposed: WT[n][k]
      int idx = t * 16 + e;
      int kk = idx >> 7, nn = idx & 127;
      WT[nn * 32 + kk] = f2bf(W[(size_t)(k0 + kk) * Ndim + nbase + nn]);
    }
    __syncthreads();
#pragma unroll
    for (int mt = 0; mt < 2; ++mt) {
      bf16x16 a = frag_a(As + (wm + mt * 16) * 32, 32);
#pragma unroll
      for (int nt = 0; nt < 4; ++nt)
        acc[mt][nt] = wmma_bf16(a, frag_b(WT + (wn + nt * 16) * 32, 32), acc[mt][nt]);
    }
    __syncthreads();
  }
#pragma unroll
  for (int mt = 0; mt < 2; ++mt)
#pragma unroll
    for (int nt = 0; nt < 4; ++nt) {
      int col = nbase + wn + nt * 16 + n15;
      float bv = bias[col];
#pragma unroll
      for (int r = 0; r < 8; ++r) {
        int row = mbase + wm + mt * 16 + r + 8 * hf;
        C[(size_t)row * Ndim + col] = f2bf(acc[mt][nt][r] + bv);
      }
    }
}

// =============================================================================
// Kernel 2: column stats of scores = qs @ ks^T (softmax over query axis)
// Per key column j: m_j = max_i s_ij, Z_j = sum_i exp(s_ij - m_j).
// block handles 128 columns (16/wave), loops all 4096 rows. bf16 tiles are
// staged with async global->LDS copies.
// =============================================================================
__global__ void shuffle_colstats(const bf16* __restrict__ qs, const bf16* __restrict__ ks,
                                 float* __restrict__ colmax, float* __restrict__ colsum) {
  extern __shared__ char smem[];
  bf16* Ks = (bf16*)smem;                  // [128][128]  32KB (resident)
  bf16* Qs = (bf16*)(smem + 128*128*2);    // [64][128]   16KB
  const int t = threadIdx.x, lane = t & 31, wid = t >> 5;
  const int b = blockIdx.y, jbase = blockIdx.x * 128;
  const uint32_t ksoff = (uint32_t)(uintptr_t)Ks;
  const uint32_t qsoff = (uint32_t)(uintptr_t)Qs;

  {   // async stage Ks: 32KB = 2048 x 16B (8 chunks/thread)
    const char* src = (const char*)(ks + (size_t)(b * SS + jbase) * SDIM);
    for (int c = t; c < 2048; c += 256) async_ld16(ksoff + c * 16, src + c * 16);
  }
  wait_async0();
  __syncthreads();

  const int jw = wid * 16;
  float m = -__builtin_inff(), z = 0.f;
  for (int i0 = 0; i0 < SS; i0 += 64) {
    {   // async stage Qs: 16KB = 1024 x 16B (4 chunks/thread)
      const char* src = (const char*)(qs + (size_t)(b * SS + i0) * SDIM);
      for (int c = t; c < 1024; c += 256) async_ld16(qsoff + c * 16, src + c * 16);
    }
    wait_async0();
    __syncthreads();
#pragma unroll
    for (int it = 0; it < 4; ++it) {
      f32x8 s = F32X8_ZERO;
#pragma unroll
      for (int kc = 0; kc < 4; ++kc)
        s = wmma_bf16(frag_a(Qs + it * 16 * SDIM + kc * 32, SDIM),
                      frag_b(Ks + jw * SDIM + kc * 32, SDIM), s);
      float tmax = s[0];
#pragma unroll
      for (int r = 1; r < 8; ++r) tmax = fmaxf(tmax, s[r]);
      float nm = fmaxf(m, tmax);
      z *= __expf(m - nm);
#pragma unroll
      for (int r = 0; r < 8; ++r) z += __expf(s[r] - nm);
      m = nm;
    }
    __syncthreads();
  }
  // combine the two lane halves (rows r+0..7 vs r+8..15 of the same column)
  float mo = __shfl_xor(m, 16, 32);
  float zo = __shfl_xor(z, 16, 32);
  float Mf = fmaxf(m, mo);
  float Zf = z * __expf(m - Mf) + zo * __expf(mo - Mf);
  if (lane < 16) {
    colmax[b * SS + jbase + jw + lane] = Mf;
    colsum[b * SS + jbase + jw + lane] = Zf;
  }
}

// =============================================================================
// Kernel 3: x1 = gamma * (exp(S - m_j)/Z_j) @ x + (1-gamma) * x   (fused)
// block: 16 query rows x full D=1024; 8 waves, 128 d-cols each; j step 32.
// Register budget: acc 64 VGPRs + q-frags 32 VGPRs -> no spills.
// =============================================================================
__global__ void shuffle_apply(const float* __restrict__ x, const bf16* __restrict__ qs,
                              const bf16* __restrict__ ks, const float* __restrict__ colmax,
                              const float* __restrict__ colsum, const float* __restrict__ gamma_p,
                              float* __restrict__ x1) {
  extern __shared__ char smem[];
  bf16* Ks = (bf16*)smem;                             // [32][128]   8KB
  bf16* XT = (bf16*)(smem + 8192);                    // [1024][32]  64KB (d-major)
  bf16* Pl = (bf16*)(smem + 8192 + 65536);            // 8 x [16][32] 8KB
  const int t = threadIdx.x, lane = t & 31, wid = t >> 5;
  const int n15 = lane & 15, hf = lane >> 4;
  const int b = blockIdx.y, ibase = blockIdx.x * 16;
  const int d0 = wid * 128;
  bf16* Pw = Pl + wid * 16 * 32;
  const uint32_t ksoff = (uint32_t)(uintptr_t)Ks;

  f32x8 acc[8];
#pragma unroll
  for (int dt = 0; dt < 8; ++dt) acc[dt] = F32X8_ZERO;

  bf16x16 aq[4];
#pragma unroll
  for (int kc = 0; kc < 4; ++kc)
    aq[kc] = frag_a(qs + (size_t)(b * SS + ibase) * SDIM + kc * 32, SDIM);

  for (int j0 = 0; j0 < SS; j0 += 32) {
    {   // async stage 32 key rows: 8KB = 512 x 16B (2 chunks/thread)
      const char* src = (const char*)(ks + (size_t)(b * SS + j0) * SDIM);
      for (int c = t; c < 512; c += 256) async_ld16(ksoff + c * 16, src + c * 16);
    }
    {   // stage x rows transposed (d-major) + f32->bf16 (VALU convert path)
      const float* xs = x + (size_t)(b * SS + j0) * DD;
      for (int i = t; i < 32 * DD; i += 256) {
        int jr = i >> 10, d = i & 1023;
        XT[d * 32 + jr] = f2bf(xs[i]);
      }
    }
    wait_async0();
    __syncthreads();
    float mj[2], rz[2];
#pragma unroll
    for (int jt = 0; jt < 2; ++jt) {
      int jc = b * SS + j0 + jt * 16 + n15;
      mj[jt] = colmax[jc];
      rz[jt] = 1.0f / colsum[jc];
    }
#pragma unroll
    for (int jt = 0; jt < 2; ++jt) {
      f32x8 s = F32X8_ZERO;
#pragma unroll
      for (int kc = 0; kc < 4; ++kc)
        s = wmma_bf16(aq[kc], frag_b(Ks + jt * 16 * SDIM + kc * 32, SDIM), s);
#pragma unroll
      for (int r = 0; r < 8; ++r) {
        float p = __expf(s[r] - mj[jt]) * rz[jt];
        Pw[(r + 8 * hf) * 32 + jt * 16 + n15] = f2bf(p);
      }
    }
    asm volatile("s_wait_dscnt 0" ::: "memory");     // P scratch RAW (wave-local)
    {
      bf16x16 ap = frag_a(Pw, 32);
#pragma unroll
      for (int dt = 0; dt < 8; ++dt)
        acc[dt] = wmma_bf16(ap, frag_b(XT + (d0 + dt * 16) * 32, 32), acc[dt]);
    }
    __syncthreads();
  }
  const float g = gamma_p[0];
#pragma unroll
  for (int dt = 0; dt < 8; ++dt)
#pragma unroll
    for (int r = 0; r < 8; ++r) {
      int row = ibase + r + 8 * hf;
      int col = d0 + dt * 16 + n15;
      size_t a = (size_t)(b * SS + row) * DD + col;
      x1[a] = g * acc[dt][r] + (1.f - g) * x[a];
    }
}

// =============================================================================
// Kernel 4: bucket multi-head attention + residual. One block per (h, nb, b).
// S = q k^T (128x128), row softmax, att = P v, out = att + x1.
// =============================================================================
__global__ void bucket_attn(const bf16* __restrict__ q2, const bf16* __restrict__ k2,
                            const bf16* __restrict__ v2, const float* __restrict__ x1,
                            float* __restrict__ out) {
  extern __shared__ char smem[];
  bf16* Qs = (bf16*)smem;                   // [128][64] 16KB
  bf16* Kb = (bf16*)(smem + 16384);         // [128][64] 16KB
  bf16* VT = (bf16*)(smem + 32768);         // [64][128] 16KB (d-major)
  bf16* Pl = (bf16*)(smem + 49152);         // 8 x [16][128] 32KB
  const int t = threadIdx.x, lane = t & 31, wid = t >> 5;
  const int n15 = lane & 15, hf = lane >> 4;
  const int hh = blockIdx.x, nb = blockIdx.y, b = blockIdx.z;
  const int g0 = b * SS + nb * BSZ;
  const int c0 = hh * HDIM;
  const uint32_t qoff = (uint32_t)(uintptr_t)Qs;
  const uint32_t koff = (uint32_t)(uintptr_t)Kb;

  // async stage q,k tiles: 1024 x 16B chunks each (8 chunks per 128B row)
  for (int c = t; c < 1024; c += 256) {
    int r = c >> 3, cu = c & 7;
    const bf16* qsrc = q2 + (size_t)(g0 + r) * DD + c0 + cu * 8;
    const bf16* ksrc = k2 + (size_t)(g0 + r) * DD + c0 + cu * 8;
    async_ld16(qoff + c * 16, qsrc);
    async_ld16(koff + c * 16, ksrc);
  }
  for (int i = t; i < 128 * 64; i += 256) {       // stage v transposed (d-major)
    int r = i >> 6, c = i & 63;
    VT[c * BSZ + r] = v2[(size_t)(g0 + r) * DD + c0 + c];
  }
  wait_async0();
  __syncthreads();

  const int i0 = wid * 16;
  bf16x16 aq[2];
#pragma unroll
  for (int kc = 0; kc < 2; ++kc) aq[kc] = frag_a(Qs + i0 * HDIM + kc * 32, HDIM);

  f32x8 s[8];
#pragma unroll
  for (int jt = 0; jt < 8; ++jt) {
    s[jt] = F32X8_ZERO;
#pragma unroll
    for (int kc = 0; kc < 2; ++kc)
      s[jt] = wmma_bf16(aq[kc], frag_b(Kb + jt * 16 * HDIM + kc * 32, HDIM), s[jt]);
  }
  // row softmax: reduce over 8 tiles in-register + 16 column lanes via shfl
#pragma unroll
  for (int r = 0; r < 8; ++r) {
    float m = s[0][r];
#pragma unroll
    for (int jt = 1; jt < 8; ++jt) m = fmaxf(m, s[jt][r]);
    for (int msk = 1; msk < 16; msk <<= 1) m = fmaxf(m, __shfl_xor(m, msk, 32));
    float sum = 0.f;
#pragma unroll
    for (int jt = 0; jt < 8; ++jt) { float e = __expf(s[jt][r] - m); s[jt][r] = e; sum += e; }
    for (int msk = 1; msk < 16; msk <<= 1) sum += __shfl_xor(sum, msk, 32);
    float rcp = 1.0f / sum;
#pragma unroll
    for (int jt = 0; jt < 8; ++jt) s[jt][r] *= rcp;
  }
  bf16* Pw = Pl + wid * 16 * BSZ;
#pragma unroll
  for (int jt = 0; jt < 8; ++jt)
#pragma unroll
    for (int r = 0; r < 8; ++r)
      Pw[(r + 8 * hf) * BSZ + jt * 16 + n15] = f2bf(s[jt][r]);
  asm volatile("s_wait_dscnt 0" ::: "memory");

  f32x8 acc[4];
#pragma unroll
  for (int dt = 0; dt < 4; ++dt) acc[dt] = F32X8_ZERO;
#pragma unroll
  for (int kc = 0; kc < 4; ++kc) {
    bf16x16 ap = frag_a(Pw + kc * 32, BSZ);
#pragma unroll
    for (int dt = 0; dt < 4; ++dt)
      acc[dt] = wmma_bf16(ap, frag_b(VT + dt * 16 * BSZ + kc * 32, BSZ), acc[dt]);
  }
#pragma unroll
  for (int dt = 0; dt < 4; ++dt)
#pragma unroll
    for (int r = 0; r < 8; ++r) {
      int row = nb * BSZ + i0 + r + 8 * hf;
      int col = c0 + dt * 16 + n15;
      size_t a = (size_t)(b * SS + row) * DD + col;
      out[a] = acc[dt][r] + x1[a];
    }
}

// =============================================================================
extern "C" void kernel_launch(void* const* d_in, const int* in_sizes, int n_in,
                              void* d_out, int out_size, void* d_ws, size_t ws_size,
                              hipStream_t stream) {
  (void)in_sizes; (void)n_in; (void)out_size; (void)ws_size;
  const float* x    = (const float*)d_in[0];
  const float* gam  = (const float*)d_in[1];
  const float* Wq_s = (const float*)d_in[2];
  const float* bq_s = (const float*)d_in[3];
  const float* Wk_s = (const float*)d_in[4];
  const float* bk_s = (const float*)d_in[5];
  const float* Wq   = (const float*)d_in[6];
  const float* bq   = (const float*)d_in[7];
  const float* Wk   = (const float*)d_in[8];
  const float* bk   = (const float*)d_in[9];
  const float* Wv   = (const float*)d_in[10];
  const float* bv   = (const float*)d_in[11];
  float* out = (float*)d_out;

  char* ws = (char*)d_ws;
  size_t off = 0;
  bf16*  qs     = (bf16*)(ws + off);  off += (size_t)MM * SDIM * 2;   //  4 MB
  bf16*  ks     = (bf16*)(ws + off);  off += (size_t)MM * SDIM * 2;   //  4 MB
  float* colmax = (float*)(ws + off); off += (size_t)MM * 4;          // 64 KB
  float* colsum = (float*)(ws + off); off += (size_t)MM * 4;          // 64 KB
  float* x1     = (float*)(ws + off); off += (size_t)MM * DD * 4;     // 64 MB
  bf16*  q2     = (bf16*)(ws + off);  off += (size_t)MM * DD * 2;     // 32 MB
  bf16*  k2     = (bf16*)(ws + off);  off += (size_t)MM * DD * 2;     // 32 MB
  bf16*  v2     = (bf16*)(ws + off);  off += (size_t)MM * DD * 2;     // 32 MB

  dim3 blk(256);

  // Shuffle projections: [16384,1024] @ [1024,128]
  gemm_bias_bf16<<<dim3(MM / 128, SDIM / 128), blk, 16384, stream>>>(x, Wq_s, bq_s, qs, MM, SDIM, DD);
  gemm_bias_bf16<<<dim3(MM / 128, SDIM / 128), blk, 16384, stream>>>(x, Wk_s, bk_s, ks, MM, SDIM, DD);

  // Per-key-column softmax stats
  shuffle_colstats<<<dim3(SS / 128, BB), blk, 32768 + 16384, stream>>>(qs, ks, colmax, colsum);

  // Fused normalized-score @ x + residual blend
  shuffle_apply<<<dim3(SS / 16, BB), blk, 8192 + 65536 + 8192, stream>>>(
      x, qs, ks, colmax, colsum, gam, x1);

  // Bucket q/k/v projections: [16384,1024] @ [1024,1024]
  gemm_bias_bf16<<<dim3(MM / 128, DD / 128), blk, 16384, stream>>>(x1, Wq, bq, q2, MM, DD, DD);
  gemm_bias_bf16<<<dim3(MM / 128, DD / 128), blk, 16384, stream>>>(x1, Wk, bk, k2, MM, DD, DD);
  gemm_bias_bf16<<<dim3(MM / 128, DD / 128), blk, 16384, stream>>>(x1, Wv, bv, v2, MM, DD, DD);

  // Bucket attention + residual
  bucket_attn<<<dim3(HH, NBUCKET, BB), blk, 49152 + 32768, stream>>>(q2, k2, v2, x1, out);
}
